// GraphTransformerGreedy_83141976916337
// MI455X (gfx1250) — matrix operand order repaired and checked by reading, not compiled, not run on hardware
//
#include <hip/hip_runtime.h>
#include <hip/hip_bf16.h>

// Problem constants (match reference)
#define BB   512
#define NN   100
#define DD   256
#define HH   8
#define DHH  32
#define LL   4
#define FFD  1024          // FF * D
#define BN   (BB * NN)     // 51200 rows
#define STEPS 200
#define NEGV (-1.0e9f)

typedef __attribute__((ext_vector_type(16))) _Float16 v16h;
typedef __attribute__((ext_vector_type(8)))  _Float16 v8h;
typedef __attribute__((ext_vector_type(8)))  float    v8f;

union HF16 { v16h v; v8h h[2]; };

static __device__ __forceinline__ v8f wmma_f16(v16h a, v16h b, v8f c) {
  // D = A(16x32 f16) * B(32x16 f16) + C(16x16 f32)
  return __builtin_amdgcn_wmma_f32_16x16x32_f16(false, a, false, b, (short)0, c,
                                                false, false);
}
static __device__ __forceinline__ v8f zero8f() {
  v8f z; for (int i = 0; i < 8; i++) z[i] = 0.f; return z;
}
static __device__ __forceinline__ v16h zero16h() {
  v16h z; for (int i = 0; i < 16; i++) z[i] = (_Float16)0; return z;
}

// ---------------------------------------------------------------------------
// f32 -> f16 conversion (weights)
// ---------------------------------------------------------------------------
__global__ void f32_to_f16_kernel(const float* __restrict__ s,
                                  _Float16* __restrict__ d, int n) {
  int i = blockIdx.x * 256 + threadIdx.x;
  if (i < n) d[i] = (_Float16)s[i];
}

// ---------------------------------------------------------------------------
// Node embedding: x = nf @ emb_w.T + emb_b ;  nf [BN,3], emb_w [256,3]
// grid = BN blocks, 256 threads (one per output dim)
// ---------------------------------------------------------------------------
__global__ void embed_kernel(const float* __restrict__ nf,
                             const float* __restrict__ ew,
                             const float* __restrict__ eb,
                             float* __restrict__ x32,
                             _Float16* __restrict__ x16) {
  int bn = blockIdx.x;
  int d  = threadIdx.x;
  const float* f = nf + (size_t)bn * 3;
  float v = eb[d] + ew[d * 3 + 0] * f[0] + ew[d * 3 + 1] * f[1] +
            ew[d * 3 + 2] * f[2];
  x32[(size_t)bn * DD + d] = v;
  x16[(size_t)bn * DD + d] = (_Float16)v;
}

// ---------------------------------------------------------------------------
// WMMA GEMM: out[M,N] = A[M,K](f16) * W[N,K](f16)^T + bias
// Block = 256 threads = 8 waves; wave w computes rows [blk.x*256+w*32, +32),
// cols [blk.y*64, +64): two 16-row A fragments share each B fragment
// (8 WMMAs per 12 b128 loads per 32-K chunk). M % 256 == 0, N % 64 == 0,
// K % 32 == 0. Fragment layouts per CDNA5 ISA 7.12.2.
// ---------------------------------------------------------------------------
template <int OUT_F16, int RELU>
__global__ __launch_bounds__(256) void gemm_wmma_kernel(
    const _Float16* __restrict__ A, const _Float16* __restrict__ W,
    const float* __restrict__ bias, float* __restrict__ outF,
    _Float16* __restrict__ outH, int N, int K) {
  const int lane = threadIdx.x & 31;
  const int wave = threadIdx.x >> 5;
  const int r15  = lane & 15;
  const int hi   = lane >> 4;
  const int row0 = blockIdx.x * 256 + wave * 32;
  const int col0 = blockIdx.y * 64;

  v8f acc[2][4];
  for (int r = 0; r < 2; r++)
    for (int t = 0; t < 4; t++) acc[r][t] = zero8f();

  const _Float16* Arow0 = A + (size_t)(row0 + r15) * K;
  const _Float16* Arow1 = A + (size_t)(row0 + 16 + r15) * K;
  for (int k0 = 0; k0 < K; k0 += 32) {
    HF16 a0, a1;
    a0.h[0] = *(const v8h*)(Arow0 + k0 + hi * 8);       // K = base..base+7
    a0.h[1] = *(const v8h*)(Arow0 + k0 + 16 + hi * 8);  // K = 16+base..23+base
    a1.h[0] = *(const v8h*)(Arow1 + k0 + hi * 8);
    a1.h[1] = *(const v8h*)(Arow1 + k0 + 16 + hi * 8);
    if (k0 + 32 < K) {  // hint: pull next A chunk toward the WGP (global_prefetch_b8)
      __builtin_prefetch(Arow0 + k0 + 32, 0, 3);
      __builtin_prefetch(Arow1 + k0 + 32, 0, 3);
    }
#pragma unroll
    for (int t = 0; t < 4; t++) {
      const _Float16* Wp =
          W + (size_t)(col0 + t * 16 + r15) * K + k0 + hi * 8;
      HF16 b;
      b.h[0] = *(const v8h*)(Wp);
      b.h[1] = *(const v8h*)(Wp + 16);
      acc[0][t] = wmma_f16(a0.v, b.v, acc[0][t]);
      acc[1][t] = wmma_f16(a1.v, b.v, acc[1][t]);
    }
  }
#pragma unroll
  for (int r = 0; r < 2; r++) {
#pragma unroll
    for (int t = 0; t < 4; t++) {
      int col  = col0 + t * 16 + r15;
      float bv = bias ? bias[col] : 0.f;
#pragma unroll
      for (int e = 0; e < 8; e++) {
        int row   = row0 + r * 16 + e + hi * 8;  // C/D: lanes 0-15 M=e, 16-31 M=8+e
        float val = acc[r][t][e] + bv;
        if (RELU) val = fmaxf(val, 0.f);
        if (OUT_F16)
          outH[(size_t)row * N + col] = (_Float16)val;
        else
          outF[(size_t)row * N + col] = val;
      }
    }
  }
}

// ---------------------------------------------------------------------------
// Attention: one wave (32 threads) per (b, h, query-tile).
// grid = (B, H, 7). q16/k16/v16/o16 are [B,N,D] f16 with d = h*32 + dh.
// ---------------------------------------------------------------------------
__global__ __launch_bounds__(32) void attention_kernel(
    const _Float16* __restrict__ q16, const _Float16* __restrict__ k16,
    const _Float16* __restrict__ v16, _Float16* __restrict__ o16) {
  __shared__ float Sl[16 * 112];   // raw scores, 16 query rows x 112 key cols
  __shared__ v8h   Pl[16 * 16];    // softmax probs f16, 16 rows x 128 cols

  const int b    = blockIdx.x;
  const int h    = blockIdx.y;
  const int qt   = blockIdx.z;
  const int lane = threadIdx.x & 31;
  const int r15  = lane & 15;
  const int hi   = lane >> 4;
  const int i0   = qt * 16;

  // ---- Q fragment (A layout, 16x32, K = DH = 32 exactly) ----
  HF16 a;
  {
    int qi = i0 + r15;
    if (qi < NN) {
      const _Float16* p = q16 + ((size_t)(b * NN + qi) * DD + h * DHH);
      a.h[0] = *(const v8h*)(p + hi * 8);
      a.h[1] = *(const v8h*)(p + 16 + hi * 8);
    } else {
      a.v = zero16h();
    }
  }

  // ---- S = Q K^T / sqrt(DH) over 7 key tiles ----
  const float inv_sqrt_dh = 0.17677669529663689f;  // 1/sqrt(32)
  for (int jt = 0; jt < 7; jt++) {
    v16h bb;
    int j = jt * 16 + r15;
    if (j < NN) {
      const _Float16* p = k16 + ((size_t)(b * NN + j) * DD + h * DHH + hi * 8);
      HF16 u;
      u.h[0] = *(const v8h*)(p);
      u.h[1] = *(const v8h*)(p + 16);
      bb = u.v;
    } else {
      bb = zero16h();
    }
    v8f s = wmma_f16(a.v, bb, zero8f());
#pragma unroll
    for (int e = 0; e < 8; e++) {
      int m = e + hi * 8;
      Sl[m * 112 + jt * 16 + r15] = s[e] * inv_sqrt_dh;
    }
  }
  __syncthreads();

  // ---- softmax per query row (lanes 0..15 each own one row) ----
  if (lane < 16) {
    int i = i0 + lane;
    _Float16* Prow = ((_Float16*)Pl) + lane * 128;
    if (i < NN) {
      const float* Srow = Sl + lane * 112;
      float mx = -3.0e38f;
      for (int j = 0; j < NN; j++) mx = fmaxf(mx, Srow[j]);
      float sum = 0.f;
      for (int j = 0; j < NN; j++) sum += __expf(Srow[j] - mx);
      float rs = 1.0f / sum;
      for (int j = 0; j < NN; j++)
        Prow[j] = (_Float16)(__expf(Srow[j] - mx) * rs);
      for (int j = NN; j < 128; j++) Prow[j] = (_Float16)0;
    } else {
      for (int j = 0; j < 128; j++) Prow[j] = (_Float16)0;
    }
  }
  __syncthreads();

  // ---- O = P V : K dim = 128 (padded keys), 4 chunks of 32 ----
  v8f acc[2];
  acc[0] = zero8f();
  acc[1] = zero8f();
  for (int kc = 0; kc < 4; kc++) {
    HF16 pa;  // A fragment from LDS P (row-major, 128 cols = 16 v8h per row)
    pa.h[0] = Pl[r15 * 16 + kc * 4 + hi];       // K = kc*32 + hi*8 ..
    pa.h[1] = Pl[r15 * 16 + kc * 4 + 2 + hi];   // K = kc*32 + 16 + hi*8 ..
#pragma unroll
    for (int t = 0; t < 2; t++) {
      v16h bb;
      int d = t * 16 + r15;
      for (int s = 0; s < 16; s++) {
        int j = kc * 32 + hi * 8 + (s & 7) + ((s >> 3) << 4);
        bb[s] = (j < NN)
                    ? v16[(size_t)(b * NN + j) * DD + h * DHH + d]
                    : (_Float16)0;
      }
      acc[t] = wmma_f16(pa.v, bb, acc[t]);
    }
  }
#pragma unroll
  for (int t = 0; t < 2; t++) {
#pragma unroll
    for (int e = 0; e < 8; e++) {
      int i = i0 + e + hi * 8;
      if (i < NN)
        o16[(size_t)(b * NN + i) * DD + h * DHH + t * 16 + r15] =
            (_Float16)acc[t][e];
    }
  }
}

// ---------------------------------------------------------------------------
// Residual + post-LayerNorm: x = LN(x + f) * g + be ; writes f32 and f16 copy
// grid = BN blocks, 256 threads (one per feature)
// ---------------------------------------------------------------------------
__global__ __launch_bounds__(256) void residual_ln_kernel(
    const float* __restrict__ xin, const float* __restrict__ f,
    const float* __restrict__ g, const float* __restrict__ be,
    float* __restrict__ xout, _Float16* __restrict__ x16out) {
  __shared__ float red[256];
  int row = blockIdx.x;
  int t   = threadIdx.x;
  size_t idx = (size_t)row * DD + t;
  float v = xin[idx] + f[idx];

  red[t] = v;
  __syncthreads();
  for (int s = 128; s > 0; s >>= 1) {
    if (t < s) red[t] += red[t + s];
    __syncthreads();
  }
  float mu = red[0] * (1.0f / DD);
  __syncthreads();
  float d = v - mu;
  red[t]  = d * d;
  __syncthreads();
  for (int s = 128; s > 0; s >>= 1) {
    if (t < s) red[t] += red[t + s];
    __syncthreads();
  }
  float var = red[0] * (1.0f / DD);
  float y   = d * rsqrtf(var + 1e-5f) * g[t] + be[t];
  xout[idx]   = y;
  x16out[idx] = (_Float16)y;
}

// ---------------------------------------------------------------------------
// Greedy decoder: one block per batch element, 200 steps in-kernel.
// ---------------------------------------------------------------------------
__global__ __launch_bounds__(256) void decode_kernel(
    const float* __restrict__ x32, const float* __restrict__ Kall,
    const float* __restrict__ aqw, const float* __restrict__ aqb,
    const float* __restrict__ demands, const float* __restrict__ caps,
    int* __restrict__ routes_out, float* __restrict__ aux_out, int max_steps) {
  __shared__ float xc[DD], qv[DD], sc[NN], dem[NN];
  __shared__ int   vis[NN];
  __shared__ int   cur_s;
  __shared__ float rem_s;

  int b = blockIdx.x;
  int t = threadIdx.x;
  if (t < NN) {
    dem[t] = demands[(size_t)b * NN + t];
    vis[t] = 0;
  }
  if (t == 0) {
    cur_s = 0;
    rem_s = caps[b];
    routes_out[(size_t)b * (STEPS + 1)] = 0;  // start at depot
    aux_out[b]      = 0.0f;                   // logp
    aux_out[BB + b] = 0.0f;                   // entropy
  }
  __syncthreads();

  for (int step = 0; step < max_steps; step++) {
    int cur = cur_s;
    xc[t] = x32[((size_t)(b * NN + cur)) * DD + t];
    __syncthreads();

    // q = xc @ aq_w.T + aq_b
    {
      float acc        = aqb[t];
      const float* wr  = aqw + (size_t)t * DD;
      for (int k = 0; k < DD; k++) acc += xc[k] * wr[k];
      qv[t] = acc;
    }
    __syncthreads();

    // scores[n] = q . K_all[b,n,:] / sqrt(D)
    if (t < NN) {
      const float* kr = Kall + ((size_t)(b * NN + t)) * DD;
      float s = 0.f;
      for (int k = 0; k < DD; k++) s += qv[k] * kr[k];
      sc[t] = s * (1.0f / 16.0f);
    }
    __syncthreads();

    if (t == 0) {
      float rem = rem_s;
      bool done = (cur == 0);
      if (done) {
        for (int n = 1; n < NN; n++)
          if (!vis[n]) { done = false; break; }
      }
      bool feas = false;
      for (int n = 1; n < NN; n++) {
        bool m = (vis[n] != 0) || (dem[n] > rem);
        if (!m) { feas = true; break; }
      }
      // argmax with masks (first index wins ties, matching jnp.argmax)
      float best = -3.0e38f;
      int   bi   = 0;
      for (int n = 0; n < NN; n++) {
        bool m  = (n == 0) ? feas : ((vis[n] != 0) || (dem[n] > rem));
        float s = m ? NEGV : sc[n];
        if (s > best) { best = s; bi = n; }
      }
      int act = done ? 0 : bi;
      bool at_depot = (act == 0);
      if (!done) {
        rem_s = at_depot ? caps[b] : (rem - dem[act]);
        if (!at_depot) vis[act] = 1;
        cur_s = act;
      }
      routes_out[(size_t)b * (STEPS + 1) + 1 + step] = act;
    }
    __syncthreads();
  }
}

// ---------------------------------------------------------------------------
// Host-side launch
// ---------------------------------------------------------------------------
extern "C" void kernel_launch(void* const* d_in, const int* in_sizes, int n_in,
                              void* d_out, int out_size, void* d_ws,
                              size_t ws_size, hipStream_t stream) {
  (void)in_sizes; (void)n_in; (void)out_size; (void)ws_size;

  // setup_inputs order; params dict flattened alphabetically (JAX pytree):
  // ak_b, ak_w, aq_b, aq_w, emb_b, emb_w, layers[...]
  const float* NF  = (const float*)d_in[0];
  const float* DEM = (const float*)d_in[1];
  const float* CAP = (const float*)d_in[2];
  const float* AKB = (const float*)d_in[3];
  const float* AKW = (const float*)d_in[4];
  const float* AQB = (const float*)d_in[5];
  const float* AQW = (const float*)d_in[6];
  const float* EMB = (const float*)d_in[7];
  const float* EMW = (const float*)d_in[8];
  // per-layer leaves (sorted): b1,b2,be1,be2,bk,bo,bq,bv,g1,g2,w1,w2,wk,wo,wq,wv
  const float* Lp[LL][16];
  for (int l = 0; l < LL; l++)
    for (int j = 0; j < 16; j++) Lp[l][j] = (const float*)d_in[9 + l * 16 + j];
  enum { iB1=0,iB2,iBE1,iBE2,iBK,iBO,iBQ,iBV,iG1,iG2,iW1,iW2,iWK,iWO,iWQ,iWV };

  // ---- workspace layout ----
  char*  ws  = (char*)d_ws;
  size_t off = 0;
  auto alloc = [&](size_t bytes) -> void* {
    void* p = ws + off;
    off += (bytes + 255) & ~(size_t)255;
    return p;
  };
  float*    x32   = (float*)alloc((size_t)BN * DD * 4);
  _Float16* x16   = (_Float16*)alloc((size_t)BN * DD * 2);
  float*    tmp32 = (float*)alloc((size_t)BN * DD * 4);  // also K_all at end
  _Float16* q16   = (_Float16*)alloc((size_t)BN * DD * 2);
  _Float16* k16   = (_Float16*)alloc((size_t)BN * DD * 2);
  _Float16* v16   = (_Float16*)alloc((size_t)BN * DD * 2);
  _Float16* o16   = (_Float16*)alloc((size_t)BN * DD * 2);
  _Float16* h16   = q16;  // FFN hidden reuses q/k/v/o region (same total size)
  _Float16* wbuf  =
      (_Float16*)alloc((size_t)(LL * (4 * DD * DD + 2 * DD * FFD) + DD * DD) * 2);

  _Float16 *wq16[LL], *wk16[LL], *wv16[LL], *wo16[LL], *w116[LL], *w216[LL];
  {
    _Float16* p = wbuf;
    for (int l = 0; l < LL; l++) {
      wq16[l] = p; p += DD * DD;
      wk16[l] = p; p += DD * DD;
      wv16[l] = p; p += DD * DD;
      wo16[l] = p; p += DD * DD;
      w116[l] = p; p += (size_t)FFD * DD;
      w216[l] = p; p += (size_t)DD * FFD;
    }
  }
  _Float16* ak16 = wbuf + (size_t)LL * (4 * DD * DD + 2 * (size_t)DD * FFD);

  auto cvt = [&](const float* s, _Float16* d, int n) {
    f32_to_f16_kernel<<<dim3((n + 255) / 256), dim3(256), 0, stream>>>(s, d, n);
  };
  for (int l = 0; l < LL; l++) {
    cvt(Lp[l][iWQ], wq16[l], DD * DD);
    cvt(Lp[l][iWK], wk16[l], DD * DD);
    cvt(Lp[l][iWV], wv16[l], DD * DD);
    cvt(Lp[l][iWO], wo16[l], DD * DD);
    cvt(Lp[l][iW1], w116[l], FFD * DD);
    cvt(Lp[l][iW2], w216[l], DD * FFD);
  }
  cvt(AKW, ak16, DD * DD);

  auto gemmH = [&](const _Float16* A, const _Float16* W, const float* bias,
                   _Float16* out, int N, int K) {
    gemm_wmma_kernel<1, 0><<<dim3(BN / 256, N / 64), 256, 0, stream>>>(
        A, W, bias, nullptr, out, N, K);
  };
  auto gemmHRelu = [&](const _Float16* A, const _Float16* W, const float* bias,
                       _Float16* out, int N, int K) {
    gemm_wmma_kernel<1, 1><<<dim3(BN / 256, N / 64), 256, 0, stream>>>(
        A, W, bias, nullptr, out, N, K);
  };
  auto gemmF = [&](const _Float16* A, const _Float16* W, const float* bias,
                   float* out, int N, int K) {
    gemm_wmma_kernel<0, 0><<<dim3(BN / 256, N / 64), 256, 0, stream>>>(
        A, W, bias, out, nullptr, N, K);
  };

  // ---- encoder ----
  embed_kernel<<<dim3(BN), dim3(256), 0, stream>>>(NF, EMW, EMB, x32, x16);

  for (int l = 0; l < LL; l++) {
    gemmH(x16, wq16[l], Lp[l][iBQ], q16, DD, DD);
    gemmH(x16, wk16[l], Lp[l][iBK], k16, DD, DD);
    gemmH(x16, wv16[l], Lp[l][iBV], v16, DD, DD);
    attention_kernel<<<dim3(BB, HH, 7), dim3(32), 0, stream>>>(q16, k16, v16,
                                                               o16);
    gemmF(o16, wo16[l], Lp[l][iBO], tmp32, DD, DD);
    residual_ln_kernel<<<dim3(BN), dim3(256), 0, stream>>>(
        x32, tmp32, Lp[l][iG1], Lp[l][iBE1], x32, x16);
    gemmHRelu(x16, w116[l], Lp[l][iB1], h16, FFD, DD);
    gemmF(h16, w216[l], Lp[l][iB2], tmp32, DD, FFD);
    residual_ln_kernel<<<dim3(BN), dim3(256), 0, stream>>>(
        x32, tmp32, Lp[l][iG2], Lp[l][iBE2], x32, x16);
  }

  // ---- decoder ----
  gemmF(x16, ak16, AKB, tmp32 /* = K_all */, DD, DD);
  decode_kernel<<<dim3(BB), dim3(256), 0, stream>>>(
      x32, tmp32, AQW, AQB, DEM, CAP, (int*)d_out,
      ((float*)d_out) + (size_t)BB * (STEPS + 1), STEPS);
}
